// OptimizedHybridAttention_584115553024
// MI455X (gfx1250) — matrix operand order repaired and checked by reading, not compiled, not run on hardware
//
#include <hip/hip_runtime.h>
#include <hip/hip_bf16.h>

// ---------------------------------------------------------------------------
// MI455X (gfx1250): all contractions via v_wmma_f32_16x16x32_bf16,
// 32x64 register-blocked tiles per wave32 (8 WMMAs / K-step).
// ---------------------------------------------------------------------------

typedef __bf16 bf16_t;
typedef __attribute__((ext_vector_type(16))) __bf16 bf16x16;
typedef __attribute__((ext_vector_type(8)))  __bf16 bf16x8;
typedef __attribute__((ext_vector_type(8)))  float  f32x8;
typedef __attribute__((ext_vector_type(4)))  float  f32x4;

static constexpr int B_       = 4;
static constexpr int S_TOK    = 4096;
static constexpr int S_STATE  = 512;
static constexpr int DMODEL   = 1024;
static constexpr int N_HEADS  = 16;
static constexpr int HEAD_DIM = 64;

// ===========================================================================
// Generic batched WMMA GEMM:  C[m,n] = sum_k A[m,k] * B[k,n]  (+ bias[n])
//   A: row-major, K contiguous (bf16, or fp32 converted on the fly)
//   B: K-major   (bf16): row k holds N contiguous values
//   wave tile: (16*TM) x (16*TN) = 32x64 ; 8 independent WMMAs per K-step
//   flat 1D wave->tile map (all shapes divide exactly; no divergence so the
//   EXEC-all-ones WMMA requirement holds)
// ===========================================================================
template <bool AF32>
__global__ void gemm_wmma_kernel(
    const void* __restrict__ Ap, long long lda, long long sAb, long long sAh,
    const bf16_t* __restrict__ Bp, long long ldb, long long sBb, long long sBh,
    bf16_t* __restrict__ Cb, long long ldcb, long long sCbB, long long sCbH,
    float* __restrict__ Cf, long long ldcf, long long sCfB, long long sCfH,
    const float* __restrict__ bias,
    int M, int N, int K, int H)
{
    constexpr int TM = 2, TN = 4;
    const int lane = threadIdx.x & 31;
    const int wave = threadIdx.x >> 5;
    const int tilesN = N / (16 * TN);
    const long long waveId = (long long)blockIdx.x * 4 + wave;
    const int n0 = (int)(waveId % tilesN) * (16 * TN);
    const int m0 = (int)(waveId / tilesN) * (16 * TM);
    const int z  = blockIdx.z;
    const int bz = z / H;
    const int hz = z % H;

    const float*  Af = nullptr;
    const bf16_t* Ab = nullptr;
    if (AF32) Af = (const float*)Ap  + (long long)bz * sAb + (long long)hz * sAh;
    else      Ab = (const bf16_t*)Ap + (long long)bz * sAb + (long long)hz * sAh;
    const bf16_t* Bb = Bp + (long long)bz * sBb + (long long)hz * sBh;

    // A fragment addressing (ISA 7.12.2, 16-bit A 16x32):
    //   lanes 0-15 : M=lane,    K = {k0+0..7, k0+16..23}
    //   lanes16-31 : M=lane-16, K = {k0+8..15, k0+24..31}
    const int mr   = lane & 15;
    const int ksel = (lane < 16) ? 0 : 8;

    f32x8 acc[TM][TN] = {};

    for (int k0 = 0; k0 < K; k0 += 32) {
        // prefetch next K-slab (speculative; safe past end of buffer)
        __builtin_prefetch((const void*)(Bb + (long long)(k0 + 32 + lane) * ldb + n0), 0, 1);

        bf16x16 af[TM];
        #pragma unroll
        for (int t = 0; t < TM; ++t) {
            const long long row = m0 + t * 16 + mr;
            if (AF32) {
                const float* ar = Af + row * lda + k0 + ksel;
                f32x4 a0 = *(const f32x4*)(ar);
                f32x4 a1 = *(const f32x4*)(ar + 4);
                f32x4 a2 = *(const f32x4*)(ar + 16);
                f32x4 a3 = *(const f32x4*)(ar + 20);
                #pragma unroll
                for (int i = 0; i < 4; ++i) {
                    af[t][i]      = (bf16_t)a0[i];
                    af[t][i + 4]  = (bf16_t)a1[i];
                    af[t][i + 8]  = (bf16_t)a2[i];
                    af[t][i + 12] = (bf16_t)a3[i];
                }
            } else {
                const bf16_t* ar = Ab + row * lda + k0 + ksel;
                __builtin_prefetch((const void*)(ar + 32), 0, 1);
                bf16x8 a0 = *(const bf16x8*)(ar);
                bf16x8 a1 = *(const bf16x8*)(ar + 16);
                #pragma unroll
                for (int i = 0; i < 8; ++i) { af[t][i] = a0[i]; af[t][i + 8] = a1[i]; }
            }
        }

        // B fragments: N = lane&15, K = {k0+ksel+0..7, k0+ksel+16..23} (same as A)
        const bf16_t* brow = Bb + (long long)(k0 + ksel) * ldb + n0 + mr;
        bf16x16 bfr[TN];
        #pragma unroll
        for (int t = 0; t < TN; ++t) {
            #pragma unroll
            for (int i = 0; i < 8; ++i) {
                bfr[t][i]     = brow[(long long)i * ldb + t * 16];
                bfr[t][i + 8] = brow[(long long)(16 + i) * ldb + t * 16];
            }
        }

        #pragma unroll
        for (int i = 0; i < TM; ++i)
            #pragma unroll
            for (int j = 0; j < TN; ++j)
                acc[i][j] = __builtin_amdgcn_wmma_f32_16x16x32_bf16(
                    false, af[i], false, bfr[j], (short)0, acc[i][j], false, false);
    }

    // C layout (ISA 7.12.2, 32-bit C/D 16x16): N = lane&15,
    //   VGPR r -> M = r (lanes 0-15) / r+8 (lanes 16-31)
    const int mbase = ((lane < 16) ? 0 : 8);
    bf16_t* cb = Cb ? (Cb + (long long)bz * sCbB + (long long)hz * sCbH) : nullptr;
    float*  cf = Cf ? (Cf + (long long)bz * sCfB + (long long)hz * sCfH) : nullptr;
    #pragma unroll
    for (int i = 0; i < TM; ++i) {
        #pragma unroll
        for (int j = 0; j < TN; ++j) {
            const int n = n0 + j * 16 + (lane & 15);
            const float bv = bias ? bias[n] : 0.0f;
            #pragma unroll
            for (int r = 0; r < 8; ++r) {
                float v = acc[i][j][r] + bv;
                long long row = m0 + i * 16 + mbase + r;
                if (cf) cf[row * ldcf + n] = v;
                if (cb) cb[row * ldcb + n] = (bf16_t)v;
            }
        }
    }
}

// ===========================================================================
// Support kernels
// ===========================================================================

// W[o][i] fp32 -> WT[i][o] bf16 (K-major B operand)
__global__ void wt_convert_kernel(const float* __restrict__ W,
                                  bf16_t* __restrict__ WT, int O, int I)
{
    long long idx = (long long)blockIdx.x * blockDim.x + threadIdx.x;
    if (idx >= (long long)O * I) return;
    int i = (int)(idx % I);
    int o = (int)(idx / I);
    WT[(long long)i * O + o] = (bf16_t)W[idx];
}

__global__ void cvt_bf16_kernel(const float* __restrict__ in,
                                bf16_t* __restrict__ out, long long n)
{
    long long i = (long long)blockIdx.x * blockDim.x + threadIdx.x;
    if (i < n) out[i] = (bf16_t)in[i];
}

// k[b][s][h*64+d] -> kt[b][h][d][s], LDS-tiled 32x32 transpose (coalesced both ways)
__global__ void transpose_heads_kernel(const bf16_t* __restrict__ in,
                                       bf16_t* __restrict__ out, int S)
{
    __shared__ bf16_t tile[32][33];
    const int z = blockIdx.z;
    const int b = z / N_HEADS;
    const int h = z % N_HEADS;
    const int s0 = blockIdx.x * 32;
    const int d0 = blockIdx.y * 32;
    const int tx = threadIdx.x;   // 0..31
    const int ty = threadIdx.y;   // 0..7
    #pragma unroll
    for (int r = 0; r < 32; r += 8) {
        int s = s0 + ty + r;
        tile[ty + r][tx] = in[((long long)b * S + s) * DMODEL + h * HEAD_DIM + d0 + tx];
    }
    __syncthreads();
    #pragma unroll
    for (int r = 0; r < 32; r += 8) {
        int d = d0 + ty + r;
        out[(((long long)b * N_HEADS + h) * HEAD_DIM + d) * S + s0 + tx] = tile[tx][ty + r];
    }
}

// LayerNorm over D=1024, optional residual, fp32 and/or bf16 outputs
__global__ void layernorm_kernel(const float* __restrict__ x,
                                 const float* __restrict__ resid,
                                 const float* __restrict__ g,
                                 const float* __restrict__ bta,
                                 float* __restrict__ outf,
                                 bf16_t* __restrict__ outb,
                                 int D)
{
    const long long row = blockIdx.x;
    const float* px = x + row * (long long)D;
    const float* pr = resid ? resid + row * (long long)D : nullptr;
    const int tid = threadIdx.x;
    const int items = D / 256;   // 4
    float v[4];
    float s = 0.f, ss = 0.f;
    for (int i = 0; i < items; ++i) {
        float t = px[i * 256 + tid];
        if (pr) t += pr[i * 256 + tid];
        v[i] = t; s += t; ss += t * t;
    }
    __shared__ float r1[256], r2[256];
    r1[tid] = s; r2[tid] = ss; __syncthreads();
    for (int k = 128; k > 0; k >>= 1) {
        if (tid < k) { r1[tid] += r1[tid + k]; r2[tid] += r2[tid + k]; }
        __syncthreads();
    }
    float mean = r1[0] / D;
    float var  = r2[0] / D - mean * mean;
    float inv  = rsqrtf(var + 1e-5f);
    for (int i = 0; i < items; ++i) {
        int c = i * 256 + tid;
        float y = (v[i] - mean) * inv * g[c] + bta[c];
        if (outf) outf[row * (long long)D + c] = y;
        if (outb) outb[row * (long long)D + c] = (bf16_t)y;
    }
}

// In-place row softmax of scale*x ; rows held in registers (L = ITEMS*256)
template <int ITEMS>
__global__ void softmax_rows_kernel(float* __restrict__ w, int L, float scale)
{
    const long long row = blockIdx.x;
    float* p = w + row * (long long)L;
    const int tid = threadIdx.x;
    float v[ITEMS];
    float m = -3.4e38f;
    #pragma unroll
    for (int i = 0; i < ITEMS; ++i) {
        v[i] = p[i * 256 + tid] * scale;
        m = fmaxf(m, v[i]);
    }
    __shared__ float red[256];
    red[tid] = m; __syncthreads();
    for (int s = 128; s > 0; s >>= 1) {
        if (tid < s) red[tid] = fmaxf(red[tid], red[tid + s]);
        __syncthreads();
    }
    m = red[0]; __syncthreads();
    float sum = 0.f;
    #pragma unroll
    for (int i = 0; i < ITEMS; ++i) { v[i] = __expf(v[i] - m); sum += v[i]; }
    red[tid] = sum; __syncthreads();
    for (int s = 128; s > 0; s >>= 1) {
        if (tid < s) red[tid] += red[tid + s];
        __syncthreads();
    }
    float inv = 1.0f / red[0];
    #pragma unroll
    for (int i = 0; i < ITEMS; ++i) p[i * 256 + tid] = v[i] * inv;
}

// gated = sigmoid(glog)*ts + (1-sigmoid)*ss -> bf16
__global__ void gate_fuse_kernel(const float* __restrict__ gl,
                                 const float* __restrict__ tsf,
                                 const float* __restrict__ ssf,
                                 bf16_t* __restrict__ gated, long long n)
{
    long long i = (long long)blockIdx.x * blockDim.x + threadIdx.x;
    if (i >= n) return;
    float g = 1.0f / (1.0f + __expf(-gl[i]));
    gated[i] = (bf16_t)(g * tsf[i] + (1.0f - g) * ssf[i]);
}

// ===========================================================================
// Host orchestration
// ===========================================================================
extern "C" void kernel_launch(void* const* d_in, const int* in_sizes, int n_in,
                              void* d_out, int out_size, void* d_ws, size_t ws_size,
                              hipStream_t stream)
{
    (void)in_sizes; (void)n_in; (void)out_size; (void)ws_size;
    const float* tokens = (const float*)d_in[0];
    const float* states = (const float*)d_in[1];
    const float* w_ts_q = (const float*)d_in[2];
    const float* w_ts_k = (const float*)d_in[3];
    const float* w_ts_v = (const float*)d_in[4];
    const float* w_ss_q = (const float*)d_in[5];
    const float* w_ss_k = (const float*)d_in[6];
    const float* w_ss_v = (const float*)d_in[7];
    const float* w_out  = (const float*)d_in[8];
    const float* b_out  = (const float*)d_in[9];
    const float* w_gate = (const float*)d_in[10];
    const float* b_gate = (const float*)d_in[11];
    const float* ln1_g  = (const float*)d_in[12];
    const float* ln1_b  = (const float*)d_in[13];
    const float* ln2_g  = (const float*)d_in[14];
    const float* ln2_b  = (const float*)d_in[15];
    const float* ln3_g  = (const float*)d_in[16];
    const float* ln3_b  = (const float*)d_in[17];

    const long long ROWS_S = (long long)B_ * S_STATE;   // 2048
    const long long ROWS_T = (long long)B_ * S_TOK;     // 16384

    float* out  = (float*)d_out;
    float* ts_w = out  + ROWS_S * DMODEL;                               // [B,16,512,4096]
    float* ss_w = ts_w + (long long)B_ * N_HEADS * S_STATE * S_TOK;     // [B,16,512,512]

    // ---- workspace carve-up (256B aligned) ----
    char* wsp = (char*)d_ws;
    auto alloc = [&](long long bytes) -> char* {
        char* p = wsp;
        wsp += (bytes + 255) & ~255LL;
        return p;
    };
    bf16_t* tok_bf = (bf16_t*)alloc(ROWS_T * DMODEL * 2);
    bf16_t* s1     = (bf16_t*)alloc(ROWS_S * DMODEL * 2);
    bf16_t* s2     = (bf16_t*)alloc(ROWS_S * DMODEL * 2);
    bf16_t* WTtq   = (bf16_t*)alloc((long long)DMODEL * DMODEL * 2);
    bf16_t* WTtk   = (bf16_t*)alloc((long long)DMODEL * DMODEL * 2);
    bf16_t* WTtv   = (bf16_t*)alloc((long long)DMODEL * DMODEL * 2);
    bf16_t* WTsq   = (bf16_t*)alloc((long long)DMODEL * DMODEL * 2);
    bf16_t* WTsk   = (bf16_t*)alloc((long long)DMODEL * DMODEL * 2);
    bf16_t* WTsv   = (bf16_t*)alloc((long long)DMODEL * DMODEL * 2);
    bf16_t* WTo    = (bf16_t*)alloc((long long)DMODEL * DMODEL * 2);
    bf16_t* WTg    = (bf16_t*)alloc((long long)(2 * DMODEL) * DMODEL * 2);
    bf16_t* q1     = (bf16_t*)alloc(ROWS_S * DMODEL * 2);
    bf16_t* kv1    = (bf16_t*)alloc(ROWS_T * DMODEL * 2);   // k1 natural, reused for v1
    bf16_t* k1t    = (bf16_t*)alloc(ROWS_T * DMODEL * 2);   // [B][16][64][4096]
    bf16_t* q2     = (bf16_t*)alloc(ROWS_S * DMODEL * 2);
    bf16_t* kv2    = (bf16_t*)alloc(ROWS_S * DMODEL * 2);   // k2 natural, reused for v2
    bf16_t* k2t    = (bf16_t*)alloc(ROWS_S * DMODEL * 2);   // [B][16][64][512]
    bf16_t* comb   = (bf16_t*)alloc(ROWS_S * 2 * DMODEL * 2);
    bf16_t* gated  = (bf16_t*)alloc(ROWS_S * DMODEL * 2);
    float*  tsf    = (float*)alloc(ROWS_S * DMODEL * 4);
    float*  ssf    = (float*)alloc(ROWS_S * DMODEL * 4);
    float*  glog   = (float*)alloc(ROWS_S * DMODEL * 4);
    float*  preout = (float*)alloc(ROWS_S * DMODEL * 4);

    const dim3 gblk(128);   // 4 waves of 32
    // waves = (M/32)*(N/64); blocks = waves/4  (exact for all shapes used)
    auto gemm_grid = [](long long M, long long N, int batches) {
        return dim3((unsigned)(((M / 32) * (N / 64)) / 4), 1, (unsigned)batches);
    };

    // ---- phase 0: precision conversion / weight transposes / LN ----
    {
        long long n = (long long)DMODEL * DMODEL;
        dim3 g((unsigned)((n + 255) / 256));
        wt_convert_kernel<<<g, 256, 0, stream>>>(w_ts_q, WTtq, DMODEL, DMODEL);
        wt_convert_kernel<<<g, 256, 0, stream>>>(w_ts_k, WTtk, DMODEL, DMODEL);
        wt_convert_kernel<<<g, 256, 0, stream>>>(w_ts_v, WTtv, DMODEL, DMODEL);
        wt_convert_kernel<<<g, 256, 0, stream>>>(w_ss_q, WTsq, DMODEL, DMODEL);
        wt_convert_kernel<<<g, 256, 0, stream>>>(w_ss_k, WTsk, DMODEL, DMODEL);
        wt_convert_kernel<<<g, 256, 0, stream>>>(w_ss_v, WTsv, DMODEL, DMODEL);
        wt_convert_kernel<<<g, 256, 0, stream>>>(w_out,  WTo,  DMODEL, DMODEL);
        long long ng = (long long)DMODEL * 2 * DMODEL;
        wt_convert_kernel<<<dim3((unsigned)((ng + 255) / 256)), 256, 0, stream>>>(
            w_gate, WTg, DMODEL, 2 * DMODEL);
        long long nt = ROWS_T * DMODEL;
        cvt_bf16_kernel<<<dim3((unsigned)((nt + 255) / 256)), 256, 0, stream>>>(
            tokens, tok_bf, nt);
        layernorm_kernel<<<dim3((unsigned)ROWS_S), 256, 0, stream>>>(
            states, nullptr, ln1_g, ln1_b, nullptr, s1, DMODEL);
        layernorm_kernel<<<dim3((unsigned)ROWS_S), 256, 0, stream>>>(
            states, nullptr, ln2_g, ln2_b, nullptr, s2, DMODEL);
    }

    // ---- phase 1: projection GEMMs (bf16 WMMA) ----
    gemm_wmma_kernel<false><<<gemm_grid(ROWS_S, DMODEL, 1), gblk, 0, stream>>>(
        (const void*)s1, DMODEL, 0, 0, WTtq, DMODEL, 0, 0,
        q1, DMODEL, 0, 0, (float*)nullptr, 0, 0, 0, nullptr,
        (int)ROWS_S, DMODEL, DMODEL, 1);
    gemm_wmma_kernel<false><<<gemm_grid(ROWS_T, DMODEL, 1), gblk, 0, stream>>>(
        (const void*)tok_bf, DMODEL, 0, 0, WTtk, DMODEL, 0, 0,
        kv1, DMODEL, 0, 0, (float*)nullptr, 0, 0, 0, nullptr,
        (int)ROWS_T, DMODEL, DMODEL, 1);
    transpose_heads_kernel<<<dim3(S_TOK / 32, 2, B_ * N_HEADS), dim3(32, 8), 0, stream>>>(
        kv1, k1t, S_TOK);
    gemm_wmma_kernel<false><<<gemm_grid(ROWS_T, DMODEL, 1), gblk, 0, stream>>>(
        (const void*)tok_bf, DMODEL, 0, 0, WTtv, DMODEL, 0, 0,
        kv1, DMODEL, 0, 0, (float*)nullptr, 0, 0, 0, nullptr,
        (int)ROWS_T, DMODEL, DMODEL, 1);
    gemm_wmma_kernel<false><<<gemm_grid(ROWS_S, DMODEL, 1), gblk, 0, stream>>>(
        (const void*)s2, DMODEL, 0, 0, WTsq, DMODEL, 0, 0,
        q2, DMODEL, 0, 0, (float*)nullptr, 0, 0, 0, nullptr,
        (int)ROWS_S, DMODEL, DMODEL, 1);
    gemm_wmma_kernel<false><<<gemm_grid(ROWS_S, DMODEL, 1), gblk, 0, stream>>>(
        (const void*)s2, DMODEL, 0, 0, WTsk, DMODEL, 0, 0,
        kv2, DMODEL, 0, 0, (float*)nullptr, 0, 0, 0, nullptr,
        (int)ROWS_S, DMODEL, DMODEL, 1);
    transpose_heads_kernel<<<dim3(S_STATE / 32, 2, B_ * N_HEADS), dim3(32, 8), 0, stream>>>(
        kv2, k2t, S_STATE);
    gemm_wmma_kernel<false><<<gemm_grid(ROWS_S, DMODEL, 1), gblk, 0, stream>>>(
        (const void*)s2, DMODEL, 0, 0, WTsv, DMODEL, 0, 0,
        kv2, DMODEL, 0, 0, (float*)nullptr, 0, 0, 0, nullptr,
        (int)ROWS_S, DMODEL, DMODEL, 1);

    // ---- phase 2: attention scores (fp32 straight into d_out) ----
    gemm_wmma_kernel<false><<<gemm_grid(S_STATE, S_TOK, B_ * N_HEADS), gblk, 0, stream>>>(
        (const void*)q1, DMODEL, (long long)S_STATE * DMODEL, HEAD_DIM,
        k1t, S_TOK, (long long)N_HEADS * HEAD_DIM * S_TOK, (long long)HEAD_DIM * S_TOK,
        (bf16_t*)nullptr, 0, 0, 0,
        ts_w, S_TOK, (long long)N_HEADS * S_STATE * S_TOK, (long long)S_STATE * S_TOK,
        nullptr, S_STATE, S_TOK, HEAD_DIM, N_HEADS);
    gemm_wmma_kernel<false><<<gemm_grid(S_STATE, S_STATE, B_ * N_HEADS), gblk, 0, stream>>>(
        (const void*)q2, DMODEL, (long long)S_STATE * DMODEL, HEAD_DIM,
        k2t, S_STATE, (long long)N_HEADS * HEAD_DIM * S_STATE, (long long)HEAD_DIM * S_STATE,
        (bf16_t*)nullptr, 0, 0, 0,
        ss_w, S_STATE, (long long)N_HEADS * S_STATE * S_STATE, (long long)S_STATE * S_STATE,
        nullptr, S_STATE, S_STATE, HEAD_DIM, N_HEADS);

    // ---- phase 3: softmax in place (scale = 1/sqrt(64)) ----
    softmax_rows_kernel<16><<<dim3((unsigned)(B_ * N_HEADS * S_STATE)), 256, 0, stream>>>(
        ts_w, S_TOK, 0.125f);
    softmax_rows_kernel<2><<<dim3((unsigned)(B_ * N_HEADS * S_STATE)), 256, 0, stream>>>(
        ss_w, S_STATE, 0.125f);

    // ---- phase 4: AV GEMMs (A = fp32 probs, converted on the fly) ----
    gemm_wmma_kernel<true><<<gemm_grid(S_STATE, HEAD_DIM, B_ * N_HEADS), gblk, 0, stream>>>(
        (const void*)ts_w, S_TOK, (long long)N_HEADS * S_STATE * S_TOK, (long long)S_STATE * S_TOK,
        kv1 /*v1*/, DMODEL, (long long)S_TOK * DMODEL, HEAD_DIM,
        comb, 2 * DMODEL, (long long)S_STATE * 2 * DMODEL, HEAD_DIM,
        tsf, DMODEL, (long long)S_STATE * DMODEL, HEAD_DIM,
        nullptr, S_STATE, HEAD_DIM, S_TOK, N_HEADS);
    gemm_wmma_kernel<true><<<gemm_grid(S_STATE, HEAD_DIM, B_ * N_HEADS), gblk, 0, stream>>>(
        (const void*)ss_w, S_STATE, (long long)N_HEADS * S_STATE * S_STATE, (long long)S_STATE * S_STATE,
        kv2 /*v2*/, DMODEL, (long long)S_STATE * DMODEL, HEAD_DIM,
        comb + DMODEL, 2 * DMODEL, (long long)S_STATE * 2 * DMODEL, HEAD_DIM,
        ssf, DMODEL, (long long)S_STATE * DMODEL, HEAD_DIM,
        nullptr, S_STATE, HEAD_DIM, S_STATE, N_HEADS);

    // ---- phase 5: gate GEMM, fuse, output projection, final LN ----
    gemm_wmma_kernel<false><<<gemm_grid(ROWS_S, DMODEL, 1), gblk, 0, stream>>>(
        (const void*)comb, 2 * DMODEL, 0, 0, WTg, DMODEL, 0, 0,
        (bf16_t*)nullptr, 0, 0, 0, glog, DMODEL, 0, 0, b_gate,
        (int)ROWS_S, DMODEL, 2 * DMODEL, 1);

    {
        long long n = ROWS_S * DMODEL;
        gate_fuse_kernel<<<dim3((unsigned)((n + 255) / 256)), 256, 0, stream>>>(
            glog, tsf, ssf, gated, n);
    }

    gemm_wmma_kernel<false><<<gemm_grid(ROWS_S, DMODEL, 1), gblk, 0, stream>>>(
        (const void*)gated, DMODEL, 0, 0, WTo, DMODEL, 0, 0,
        (bf16_t*)nullptr, 0, 0, 0, preout, DMODEL, 0, 0, b_out,
        (int)ROWS_S, DMODEL, DMODEL, 1);

    layernorm_kernel<<<dim3((unsigned)ROWS_S), 256, 0, stream>>>(
        preout, states, ln3_g, ln3_b, out, nullptr, DMODEL);
}
